// MambaBlock_23957327577425
// MI455X (gfx1250) — matrix operand order repaired and checked by reading, not compiled, not run on hardware
//
#include <hip/hip_runtime.h>
#include <hip/hip_bf16.h>
#include <math.h>

// ---------------------------------------------------------------------------
// Mamba block forward for gfx1250 (MI455X).
// GEMMs on the matrix pipe via v_wmma_f32_16x16x32_f16 (f16 in, f32 acc),
// double-buffered LDS staging, wide (b128) global and LDS accesses.
// ---------------------------------------------------------------------------

typedef __attribute__((ext_vector_type(16))) _Float16 v16h;
typedef __attribute__((ext_vector_type(8)))  _Float16 v8h;
typedef __attribute__((ext_vector_type(4)))  _Float16 v4h;
typedef __attribute__((ext_vector_type(8)))  float    v8f;

#define D_MODEL  1024
#define D_INNER  2048
#define D_STATE  16
#define DT_RANK  64
#define BATCH    4
#define SEQ      4096
#define NTOK     (BATCH * SEQ)      // 16384 rows for every GEMM

// ---------------------------------------------------------------------------
// Generic row-major GEMM:  C[M,N] = A[M,K] * W[N,K]^T   (f32 io, f16 compute)
//   block tile 128(M) x 64(N) x 32(K), 256 threads = 8 waves,
//   wave w computes rows [128*by + 16w, +16) x the block's 64 cols (4 WMMAs).
//   grid = (ceil(N/64), M/128).  M % 128 == 0, K % 32 == 0 assumed.
// WMMA fragment layouts per CDNA5 ISA 7.12.2 (wave32):
//   A 16x32 f16 : lanes 0-15 -> M=lane,    K in {0..7,16..23}
//                 lanes 16-31-> M=lane-16, K in {8..15,24..31}
//     -> with LDS [m][k], a lane's fragment is two contiguous 16B chunks.
//   B 32x16 f16 : lanes 0-15 -> N=lane, K in {0..7,16..23} ; lanes 16-31 -> K in {8..15,24..31}
//     -> with LDS [n][k] (transposed), a fragment is two contiguous 16B chunks.
//   C/D 16x16 f32: VGPR r -> M = r (lanes 0-15) / 8+r (lanes 16-31), N = lane%16
// ---------------------------------------------------------------------------
#define ASTR 40          // padded LDS row stride in halves (80 B -> bank spread)
#define BSTR 40

__global__ __launch_bounds__(256)
void gemm_wmma_f16(const float* __restrict__ A, int lda,
                   const float* __restrict__ W,
                   float* __restrict__ C, int ldc, int K, int N)
{
    __shared__ _Float16 As[2][128 * ASTR];   // [buf][m][k]
    __shared__ _Float16 Bs[2][64 * BSTR];    // [buf][n][k]

    const int tid   = threadIdx.x;
    const int wave  = tid >> 5;
    const int lane  = tid & 31;
    const int lhalf = lane & 15;
    const int hi    = lane >> 4;             // 0: lanes 0-15, 1: lanes 16-31

    const int m0 = blockIdx.y * 128;
    const int n0 = blockIdx.x * 64;

    v8f acc[4] = {{}, {}, {}, {}};

    float4 aReg[4];                          // staged A tile (128x32 = 1024 f4)
    float4 bReg[2];                          // staged B tile (64x32  = 512 f4)

    const int T = K >> 5;                    // number of 32-wide K steps

    // ---- global -> regs for K-step kt ------------------------------------
    auto gload = [&](int kt) {
        const int kb = kt * 32;
        #pragma unroll
        for (int e = 0; e < 4; ++e) {
            int q = tid + e * 256;           // 0..1023
            int r = q >> 3, c4 = q & 7;      // 8 float4 per 32-wide row
            aReg[e] = *(const float4*)&A[(size_t)(m0 + r) * lda + kb + c4 * 4];
        }
        #pragma unroll
        for (int e = 0; e < 2; ++e) {
            int q = tid + e * 256;           // 0..511
            int n = q >> 3, k4 = q & 7;
            int gn = n0 + n;
            bReg[e] = (gn < N) ? *(const float4*)&W[(size_t)gn * K + kb + k4 * 4]
                               : float4{0.f, 0.f, 0.f, 0.f};
        }
    };

    // ---- regs -> LDS (f32 -> f16, b64 stores) ----------------------------
    auto lstore = [&](int buf) {
        #pragma unroll
        for (int e = 0; e < 4; ++e) {
            int q = tid + e * 256;
            int r = q >> 3, c4 = q & 7;
            v4h h;
            h[0] = (_Float16)aReg[e].x; h[1] = (_Float16)aReg[e].y;
            h[2] = (_Float16)aReg[e].z; h[3] = (_Float16)aReg[e].w;
            *(v4h*)&As[buf][r * ASTR + c4 * 4] = h;
        }
        #pragma unroll
        for (int e = 0; e < 2; ++e) {
            int q = tid + e * 256;
            int n = q >> 3, k4 = q & 7;
            v4h h;
            h[0] = (_Float16)bReg[e].x; h[1] = (_Float16)bReg[e].y;
            h[2] = (_Float16)bReg[e].z; h[3] = (_Float16)bReg[e].w;
            *(v4h*)&Bs[buf][n * BSTR + k4 * 4] = h;
        }
    };

    gload(0);
    lstore(0);
    __syncthreads();

    const int arow = wave * 16 + lhalf;
    const int akhi = hi * 8;                 // A: +8 k for high lanes
    const int bkhi = hi * 8;                 // B: +8 k for high lanes

    for (int kt = 0; kt < T; ++kt) {
        const int cur = kt & 1;
        if (kt + 1 < T) gload(kt + 1);

        // ---- issue ALL fragment loads first (wide b128 LDS reads), so a
        // ---- single dscnt wait covers them and the 4 WMMAs go back-to-back
        v8h a_lo = *(const v8h*)&As[cur][arow * ASTR + akhi];
        v8h a_hi = *(const v8h*)&As[cur][arow * ASTR + 16 + akhi];
        v8h b_lo[4], b_hi[4];
        #pragma unroll
        for (int j = 0; j < 4; ++j) {
            const int brow = j * 16 + lhalf;
            b_lo[j] = *(const v8h*)&Bs[cur][brow * BSTR + bkhi];
            b_hi[j] = *(const v8h*)&Bs[cur][brow * BSTR + 16 + bkhi];
        }

        v16h a;
        #pragma unroll
        for (int i = 0; i < 8; ++i) { a[i] = a_lo[i]; a[8 + i] = a_hi[i]; }

        #pragma unroll
        for (int j = 0; j < 4; ++j) {
            v16h b;
            #pragma unroll
            for (int i = 0; i < 8; ++i) { b[i] = b_lo[j][i]; b[8 + i] = b_hi[j][i]; }
            acc[j] = __builtin_amdgcn_wmma_f32_16x16x32_f16(false, a, false, b,
                                                            (short)0, acc[j],
                                                            false, false);
        }

        if (kt + 1 < T) lstore((kt + 1) & 1);
        __syncthreads();
    }

    // ---- store C ---------------------------------------------------------
    const int rowbase = m0 + wave * 16 + hi * 8;
    #pragma unroll
    for (int j = 0; j < 4; ++j) {
        int col = n0 + j * 16 + lhalf;
        if (col < N) {
            #pragma unroll
            for (int r = 0; r < 8; ++r)
                C[(size_t)(rowbase + r) * ldc + col] = acc[j][r];
        }
    }
}

// ---------------------------------------------------------------------------
// Causal depthwise conv1d (width 4) + bias + SiLU.  u = xz[:, :, :2048].
// One thread per (token, channel).
// ---------------------------------------------------------------------------
__global__ __launch_bounds__(256)
void conv_silu_kernel(const float* __restrict__ xz,
                      const float* __restrict__ cw,
                      const float* __restrict__ cb,
                      float* __restrict__ uc)
{
    int idx = blockIdx.x * 256 + threadIdx.x;        // < NTOK * D_INNER
    int d = idx & (D_INNER - 1);
    int m = idx >> 11;
    int l = m & (SEQ - 1);
    float acc = cb[d];
    #pragma unroll
    for (int j = 0; j < 4; ++j) {
        int ll = l - 3 + j;
        if (ll >= 0)
            acc = fmaf(cw[d * 4 + j], xz[(size_t)(m - 3 + j) * (2 * D_INNER) + d], acc);
    }
    uc[idx] = acc / (1.f + __expf(-acc));            // SiLU
}

// ---------------------------------------------------------------------------
// delta = softplus(dt_raw + dt_proj_b)  (in place)
// ---------------------------------------------------------------------------
__global__ __launch_bounds__(256)
void softplus_kernel(float* __restrict__ delta, const float* __restrict__ bias)
{
    int idx = blockIdx.x * 256 + threadIdx.x;
    int d = idx & (D_INNER - 1);
    float v = delta[idx] + bias[d];
    delta[idx] = (v > 20.f) ? v : log1pf(__expf(v));
}

// ---------------------------------------------------------------------------
// Selective scan over L, fused with D-skip and SiLU(z) gate.
// One thread per (b, d); 16 SSM states in registers (ILP across the
// v_exp_f32 / fma chain).  B/C rows read as float4 (global_load_b128,
// broadcast across the block, L2-resident: x_dbl is only 6 MB).
// Writes gated y in place over u_c.
// ---------------------------------------------------------------------------
__global__ __launch_bounds__(256)
void scan_kernel(const float* __restrict__ xz,     // for z gate
                 float* __restrict__ uc,           // in: conv(u); out: gated y
                 const float* __restrict__ xdbl,   // [NTOK, 96] -> B at 64, C at 80
                 const float* __restrict__ delta,
                 const float* __restrict__ A_log,
                 const float* __restrict__ Dp)
{
    int idx = blockIdx.x * 256 + threadIdx.x;      // < BATCH * D_INNER
    int d = idx & (D_INNER - 1);
    int b = idx >> 11;

    float Av[D_STATE], h[D_STATE];
    #pragma unroll
    for (int n = 0; n < D_STATE; ++n) {
        Av[n] = -__expf(A_log[d * D_STATE + n]);
        h[n] = 0.f;
    }
    const float Dd = Dp[d];

    for (int l = 0; l < SEQ; ++l) {
        size_t m = (size_t)b * SEQ + l;
        float dt = delta[m * D_INNER + d];
        float u  = uc[m * D_INNER + d];
        // B row (16 f32) and C row (16 f32), each as 4x float4
        const float4* bc4 = (const float4*)(xdbl + m * 96 + DT_RANK);
        float4 Bv[4], Cv[4];
        #pragma unroll
        for (int q = 0; q < 4; ++q) { Bv[q] = bc4[q]; Cv[q] = bc4[4 + q]; }
        const float* Bf = (const float*)Bv;
        const float* Cf = (const float*)Cv;

        float du = dt * u;
        float y = 0.f;
        #pragma unroll
        for (int n = 0; n < D_STATE; ++n) {
            float dA = __expf(dt * Av[n]);
            h[n] = fmaf(dA, h[n], du * Bf[n]);
            y = fmaf(h[n], Cf[n], y);
        }
        float z = xz[m * (2 * D_INNER) + D_INNER + d];
        float g = z / (1.f + __expf(-z));          // SiLU(z)
        uc[m * D_INNER + d] = (y + u * Dd) * g;
    }
}

// ---------------------------------------------------------------------------
// LayerNorm over last dim (1024).  One block (256 threads) per token.
// ---------------------------------------------------------------------------
__global__ __launch_bounds__(256)
void layernorm_kernel(const float* __restrict__ in,
                      const float* __restrict__ w,
                      const float* __restrict__ bsc,
                      float* __restrict__ out)
{
    __shared__ float s1[256], s2[256];
    const int m = blockIdx.x;
    const int tid = threadIdx.x;
    const float* row = in + (size_t)m * D_MODEL;

    float v[4], a0 = 0.f, a1 = 0.f;
    #pragma unroll
    for (int i = 0; i < 4; ++i) {
        v[i] = row[tid + i * 256];
        a0 += v[i];
        a1 = fmaf(v[i], v[i], a1);
    }
    s1[tid] = a0; s2[tid] = a1;
    __syncthreads();
    for (int off = 128; off > 0; off >>= 1) {
        if (tid < off) { s1[tid] += s1[tid + off]; s2[tid] += s2[tid + off]; }
        __syncthreads();
    }
    float mu  = s1[0] * (1.f / D_MODEL);
    float var = s2[0] * (1.f / D_MODEL) - mu * mu;
    float rs  = rsqrtf(var + 1e-5f);
    #pragma unroll
    for (int i = 0; i < 4; ++i) {
        int c = tid + i * 256;
        out[(size_t)m * D_MODEL + c] = (v[i] - mu) * rs * w[c] + bsc[c];
    }
}

// ---------------------------------------------------------------------------
extern "C" void kernel_launch(void* const* d_in, const int* in_sizes, int n_in,
                              void* d_out, int out_size, void* d_ws, size_t ws_size,
                              hipStream_t stream)
{
    const float* x         = (const float*)d_in[0];   // (4, 4096, 1024)
    const float* in_proj_w = (const float*)d_in[1];   // (4096, 1024)
    const float* conv_w    = (const float*)d_in[2];   // (2048, 1, 4)
    const float* conv_b    = (const float*)d_in[3];   // (2048,)
    const float* x_proj_w  = (const float*)d_in[4];   // (96, 2048)
    const float* dt_proj_w = (const float*)d_in[5];   // (2048, 64)
    const float* dt_proj_b = (const float*)d_in[6];   // (2048,)
    const float* A_log     = (const float*)d_in[7];   // (2048, 16)
    const float* Dp        = (const float*)d_in[8];   // (2048,)
    const float* out_proj_w= (const float*)d_in[9];   // (1024, 2048)
    const float* ln_w      = (const float*)d_in[10];  // (1024,)
    const float* ln_b      = (const float*)d_in[11];  // (1024,)
    float* out = (float*)d_out;                       // (4, 4096, 1024)

    // workspace layout (floats)
    float* xz    = (float*)d_ws;                          // NTOK * 4096
    float* uc    = xz    + (size_t)NTOK * (2 * D_INNER);  // NTOK * 2048
    float* xdbl  = uc    + (size_t)NTOK * D_INNER;        // NTOK * 96
    float* delta = xdbl  + (size_t)NTOK * 96;             // NTOK * 2048
    float* preln = delta + (size_t)NTOK * D_INNER;        // NTOK * 1024

    const dim3 blk256(256);

    // 1) in_proj: xz = x @ in_proj_w^T   (16384 x 4096 x 1024)
    gemm_wmma_f16<<<dim3((2 * D_INNER) / 64, NTOK / 128), blk256, 0, stream>>>(
        x, D_MODEL, in_proj_w, xz, 2 * D_INNER, D_MODEL, 2 * D_INNER);

    // 2) causal depthwise conv + SiLU -> u_c
    conv_silu_kernel<<<(NTOK * D_INNER) / 256, blk256, 0, stream>>>(
        xz, conv_w, conv_b, uc);

    // 3) x_proj: x_dbl = u_c @ x_proj_w^T  (16384 x 96 x 2048)
    gemm_wmma_f16<<<dim3(2 /*ceil(96/64)*/, NTOK / 128), blk256, 0, stream>>>(
        uc, D_INNER, x_proj_w, xdbl, 96, D_INNER, 96);

    // 4) dt_proj: delta_raw = x_dbl[:, :64] @ dt_proj_w^T  (16384 x 2048 x 64)
    gemm_wmma_f16<<<dim3(D_INNER / 64, NTOK / 128), blk256, 0, stream>>>(
        xdbl, 96, dt_proj_w, delta, D_INNER, DT_RANK, D_INNER);

    // 5) delta = softplus(delta_raw + dt_proj_b)
    softplus_kernel<<<(NTOK * D_INNER) / 256, blk256, 0, stream>>>(delta, dt_proj_b);

    // 6) selective scan + D skip + SiLU(z) gate  (gated y written over u_c)
    scan_kernel<<<(BATCH * D_INNER) / 256, blk256, 0, stream>>>(
        xz, uc, xdbl, delta, A_log, Dp);

    // 7) out_proj: preln = y_gated @ out_proj_w^T  (16384 x 1024 x 2048)
    gemm_wmma_f16<<<dim3(D_MODEL / 64, NTOK / 128), blk256, 0, stream>>>(
        uc, D_INNER, out_proj_w, preln, D_MODEL, D_INNER, D_MODEL);

    // 8) LayerNorm -> out
    layernorm_kernel<<<NTOK, blk256, 0, stream>>>(preln, ln_w, ln_b, out);
}
